// ParameterizedAdjDense_412316860737
// MI455X (gfx1250) — matrix operand order repaired and checked by reading, not compiled, not run on hardware
//
#include <hip/hip_runtime.h>
#include <hip/hip_bf16.h>

// ---------------------------------------------------------------------------
// Sparse-masked SpMM: out[i] = sum_{(i,j) in edges} weight[i,j] * x[j]
// N=8192 nodes, F=256 features, E=262144 edges (unsorted COO).
//
// Bin edges by 16-row destination tile (512 tiles), then per tile reduce with
// v_wmma_f32_16x16x32_f16:
//   A (16x32 f16): column k = edge k's weight at its local dest row
//   B (32x16 f16): row    k = edge k's gathered x[col_k] 16-col slice
//   C (16x16 f32) accumulates across 32-edge groups -> masked SpMM exactly.
// No float atomics anywhere; out written exactly once. Inner loop is fully
// branch-free (sentinel-padded LDS), so EXEC stays all-ones around WMMA and
// no slow guarded tail path is generated.
// ---------------------------------------------------------------------------

#define N_NODES 8192
#define FDIM    256
#define NEDGE   (N_NODES * 32)
#define NTILES  (N_NODES / 16)   // 512 dest tiles
#define CHUNK   256              // edges staged in LDS per block iteration

typedef __attribute__((ext_vector_type(16))) _Float16 v16h;
typedef __attribute__((ext_vector_type(2)))  _Float16 v2h;
typedef __attribute__((ext_vector_type(8)))  float    v8f;

#define PK_SENTINEL (16 << 16)   // lr=16: matches no lane (m in 0..15); c=0: safe gather

// --- pass 0: zero the 512 tile counters and 512 scatter cursors -------------
__global__ void init_kernel(int* __restrict__ counts, int* __restrict__ cursor) {
    int i = threadIdx.x;
    if (i < NTILES) { counts[i] = 0; cursor[i] = 0; }
}

// --- pass 1: x (f32, 8MB) -> xh (f16, 4MB, stays L2-resident) ---------------
__global__ void cvt_x_kernel(const float* __restrict__ x, _Float16* __restrict__ xh, int n) {
    int i = blockIdx.x * blockDim.x + threadIdx.x;
    if (i < n) xh[i] = (_Float16)x[i];
}

// --- pass 2: histogram edges into dest-row tiles ----------------------------
__global__ void count_kernel(const int* __restrict__ rows, int* __restrict__ counts, int e_total) {
    int e = blockIdx.x * blockDim.x + threadIdx.x;
    if (e < e_total) atomicAdd(&counts[rows[e] >> 4], 1);
}

// --- pass 3: exclusive scan of 512 counts (single block, Hillis-Steele) -----
__global__ void scan_kernel(const int* __restrict__ counts, int* __restrict__ offsets) {
    __shared__ int s[NTILES];
    int t = threadIdx.x;
    s[t] = counts[t];
    __syncthreads();
    for (int d = 1; d < NTILES; d <<= 1) {
        int v = (t >= d) ? s[t - d] : 0;
        __syncthreads();
        s[t] += v;
        __syncthreads();
    }
    offsets[t + 1] = s[t];
    if (t == 0) offsets[0] = 0;
}

// --- pass 4: gather weight[r,c] (the only HBM-random access) and bin edges --
__global__ void scatter_kernel(const int* __restrict__ rows, const int* __restrict__ cols,
                               const float* __restrict__ weight,
                               const int* __restrict__ offsets, int* __restrict__ cursor,
                               int* __restrict__ pk, _Float16* __restrict__ wv, int e_total) {
    int e = blockIdx.x * blockDim.x + threadIdx.x;
    if (e < e_total) {
        int r = rows[e], c = cols[e];
        int t = r >> 4;
        float w = weight[(size_t)r * N_NODES + c];
        int pos = atomicAdd(&cursor[t], 1);
        int j = offsets[t] + pos;
        pk[j] = ((r & 15) << 16) | c;   // local dest row (hi) | source col (lo)
        wv[j] = (_Float16)w;
    }
}

// --- pass 5: per-tile WMMA reduction ----------------------------------------
// Block = 512 threads = 16 waves; blockIdx = dest tile; wave = 16-col slab.
// Tile's edge list staged in LDS in CHUNK-sized pieces (+32 sentinel pad),
// shared by all 16 waves.
__global__ __launch_bounds__(512) void spmm_wmma_kernel(
    const _Float16* __restrict__ xh,
    const int* __restrict__ offsets,
    const int* __restrict__ pk,
    const _Float16* __restrict__ wv,
    float* __restrict__ out) {
    __shared__ __align__(16) int      s_pk[CHUNK + 32];
    __shared__ __align__(16) _Float16 s_w [CHUNK + 32];

    const int tile  = blockIdx.x;
    const int tid   = threadIdx.x;
    const int ntile = tid >> 5;       // which 16-column slab (0..15)
    const int lane  = tid & 31;
    const int n     = lane & 15;      // N index (and M index for A build)
    const int g     = lane >> 4;      // K-group half (ISA 16-bit layouts)
    const int ncol  = ntile * 16 + n;

    const int beg = offsets[tile];
    const int end = offsets[tile + 1];

    v8f acc = {};  // 16x16 f32 accumulator (C/D), 8 VGPRs

    for (int base = beg; base < end; base += CHUNK) {
        const int cnt = min(CHUNK, end - base);
        __syncthreads();
        for (int i = tid; i < cnt; i += 512) { s_pk[i] = pk[base + i]; s_w[i] = wv[base + i]; }
        if (tid < 32) {                       // sentinel pad: zero A column, col 0 for B
            s_pk[cnt + tid] = PK_SENTINEL;
            s_w [cnt + tid] = (_Float16)0.0f;
        }
        __syncthreads();

        if (base + CHUNK < end) {             // prefetch next chunk behind the WMMA loop
            __builtin_prefetch(pk + base + CHUNK, 0, 0);
            __builtin_prefetch(wv + base + CHUNK, 0, 0);
        }

        for (int p = 0; p < cnt; p += 32) {   // 32 edges -> one WMMA, branch-free
            v16h a, b;
#pragma unroll
            for (int j = 0; j < 8; ++j) {
                const int idx = 2 * j;
                // A layout (16-bit 16x32): element idx -> K = idx + 8g (+8 if idx>=8);
                // pairs (idx, idx+1) are K-adjacent -> 8B/4B aligned paired LDS loads.
                const int  ka  = idx + 8 * g + ((idx & 8) ? 8 : 0);
                const int2 pka = *(const int2*)(&s_pk[p + ka]);
                const v2h  wp  = __builtin_bit_cast(v2h, *(const int*)(&s_w[p + ka]));
                a[idx]     = ((pka.x >> 16) == n) ? wp[0] : (_Float16)0.0f;
                a[idx + 1] = ((pka.y >> 16) == n) ? wp[1] : (_Float16)0.0f;

                // B layout (16-bit 32x16): element idx -> same K as A (ka, pair-adjacent)
                const int2 pkb = *(const int2*)(&s_pk[p + ka]);
                b[idx]     = xh[(size_t)(pkb.x & 0xFFFF) * FDIM + ncol];
                b[idx + 1] = xh[(size_t)(pkb.y & 0xFFFF) * FDIM + ncol];
            }
            // D = A*B + C  (f16 inputs, f32 accumulate)
            acc = __builtin_amdgcn_wmma_f32_16x16x32_f16(
                false, a, false, b, (short)0, acc, false, false);
        }
    }

    // C/D layout: VGPR v, lanes 0-15 -> M=v, lanes 16-31 -> M=v+8
#pragma unroll
    for (int v = 0; v < 8; ++v) {
        const int row = tile * 16 + v + 8 * g;
        out[(size_t)row * FDIM + ncol] = acc[v];
    }
}

// ---------------------------------------------------------------------------
extern "C" void kernel_launch(void* const* d_in, const int* in_sizes, int n_in,
                              void* d_out, int out_size, void* d_ws, size_t ws_size,
                              hipStream_t stream) {
    const float* x         = (const float*)d_in[0];   // [N,F]
    const float* weight    = (const float*)d_in[1];   // [N,N]
    const int*   edge_rows = (const int*)d_in[2];     // [E]
    const int*   edge_cols = (const int*)d_in[3];     // [E]
    float*       out       = (float*)d_out;           // [N,F]

    // Workspace layout (~5.6 MB total)
    char* ws = (char*)d_ws;
    int*      counts  = (int*)(ws + 0);                               // 512 ints
    int*      cursor  = (int*)(ws + 2048);                            // 512 ints
    int*      offsets = (int*)(ws + 4096);                            // 513 ints
    _Float16* xh      = (_Float16*)(ws + 8192);                       // N*F f16 (4 MB)
    int*      pkv     = (int*)(ws + 8192 + (size_t)N_NODES * FDIM * 2);   // E ints (1 MB)
    _Float16* wv      = (_Float16*)((char*)pkv + (size_t)NEDGE * 4);      // E f16 (0.5 MB)

    init_kernel<<<1, NTILES, 0, stream>>>(counts, cursor);
    cvt_x_kernel<<<(N_NODES * FDIM) / 256, 256, 0, stream>>>(x, xh, N_NODES * FDIM);
    count_kernel<<<NEDGE / 256, 256, 0, stream>>>(edge_rows, counts, NEDGE);
    scan_kernel<<<1, NTILES, 0, stream>>>(counts, offsets);
    scatter_kernel<<<NEDGE / 256, 256, 0, stream>>>(edge_rows, edge_cols, weight,
                                                    offsets, cursor, pkv, wv, NEDGE);
    spmm_wmma_kernel<<<NTILES, 512, 0, stream>>>(xh, offsets, pkv, wv, out);
}
